// DeformableAttention_9620726743866
// MI455X (gfx1250) — matrix operand order repaired and checked by reading, not compiled, not run on hardware
//
#include <hip/hip_runtime.h>
#include <hip/hip_bf16.h>
#include <math.h>

// ---- problem constants (from reference) ----
#define D_MODEL 256
#define NHEAD   8
#define HDIM    32
#define NLVL    4
#define NPT     4
#define NTOTAL  21760
#define BATCH   8
#define NQ      1000
#define NROWS   (NQ * BATCH)      // 8000 (row = qi*B + b)

// ---- WMMA GEMM tiling ----
#define WAVES   8
#define BM      128               // 8 waves x 16 rows
#define BN      32                // 2 x 16-col tiles per wave
#define BK      64
#define APADH   72                // f16 row pitch: 64 + 8 pad (144B rows: b128-aligned, conflict-free)

typedef _Float16 half_t;
typedef __attribute__((ext_vector_type(4)))  float   v4f;
typedef __attribute__((ext_vector_type(8)))  float   v8f;
typedef __attribute__((ext_vector_type(4)))  _Float16 v4h;
typedef __attribute__((ext_vector_type(8)))  _Float16 v8h;
typedef __attribute__((ext_vector_type(16))) _Float16 v16h;

#define SHUF16(a, b) __builtin_shufflevector(a, b, 0,1,2,3,4,5,6,7,8,9,10,11,12,13,14,15)

// C[row, tileCol+0..31] = A[row,:] @ W[:,tileCol+0..31] + bias.
// fp32 inputs are split into hi/lo f16 during LDS staging; product computed as
// Ah*Wh + Ah*Wl + Al*Wh with fp32 WMMA accumulation (~fp32 accuracy).
__global__ __launch_bounds__(256) void msda_wmma_gemm_f16split(
    const float* __restrict__ A, long strideA, long batchStrideA,
    const float* __restrict__ W, int ldw,
    const float* __restrict__ bias,
    float* __restrict__ C, long strideC, long batchStrideC,
    int M, int K)
{
    __shared__ half_t AhL[BM * APADH];          // 18 KB
    __shared__ half_t AlL[BM * APADH];          // 18 KB
    __shared__ half_t WthL[BN * APADH];         // transposed [n][k], 4.5 KB
    __shared__ half_t WtlL[BN * APADH];         // 4.5 KB

    const int tid     = threadIdx.x;
    const int wave    = tid >> 5;
    const int lane    = tid & 31;
    const int half    = lane >> 4;     // half-wave selector (K-range / M+8)
    const int colLane = lane & 15;     // N col within 16-tile / A row within wave tile
    const int tileCol = blockIdx.x * BN;
    const int rowBase = blockIdx.y * BM;

    const float* Ab = A + (long)blockIdx.z * batchStrideA;
    float*       Cb = C + (long)blockIdx.z * batchStrideC;

    v8f acc0 = {}, acc1 = {};

    for (int kb = 0; kb < K; kb += BK) {
        // ---- stage A panel (BM x BK fp32 -> hi/lo f16), float4 global reads ----
        for (int i = tid; i < BM * (BK / 4); i += 256) {
            int r  = i >> 4;                     // 16 float4 per row
            int c4 = (i & 15) << 2;
            int grow = rowBase + r;
            v4f val = {};
            if (grow < M)
                val = *(const v4f*)(Ab + (long)grow * strideA + kb + c4);
            v4h hi, lo;
            #pragma unroll
            for (int j = 0; j < 4; j++) {
                float  x = val[j];
                half_t h = (half_t)x;
                hi[j] = h;
                lo[j] = (half_t)(x - (float)h);
            }
            *(v4h*)(&AhL[r * APADH + c4]) = hi;
            *(v4h*)(&AlL[r * APADH + c4]) = lo;
        }
        // ---- stage W panel (BK x BN fp32 -> hi/lo f16, TRANSPOSED to [n][k]) ----
        for (int i = tid; i < BK * (BN / 4); i += 256) {
            int r  = i >> 3;                     // 8 float4 per 32-wide row
            int c4 = (i & 7) << 2;
            v4f val = *(const v4f*)(W + (long)(kb + r) * ldw + tileCol + c4);
            #pragma unroll
            for (int j = 0; j < 4; j++) {
                float  x = val[j];
                half_t h = (half_t)x;
                WthL[(c4 + j) * APADH + r] = h;
                WtlL[(c4 + j) * APADH + r] = (half_t)(x - (float)h);
            }
        }
        __syncthreads();

        // ---- v_wmma_f32_16x16x32_f16: 2 K-steps x 2 col tiles x 3 split terms ----
        // A frag (16x32 f16): lane m=lane&15 holds K = 16*half+0..15
        // B frag (32x16 f16): lane n=lane&15 holds K = 16*half+0..15 (contiguous in Wt)
        const half_t* ArH = &AhL[(wave * 16 + colLane) * APADH];
        const half_t* ArL = &AlL[(wave * 16 + colLane) * APADH];
        #pragma unroll
        for (int kk = 0; kk < BK; kk += 32) {
            const int ka = kk + 16 * half;
            v16h ah = SHUF16(*(const v8h*)(ArH + ka), *(const v8h*)(ArH + ka + 8));
            v16h al = SHUF16(*(const v8h*)(ArL + ka), *(const v8h*)(ArL + ka + 8));
            const int kw = kk + 16 * half;
            {   // col tile 0
                const half_t* WrH = &WthL[colLane * APADH];
                const half_t* WrL = &WtlL[colLane * APADH];
                v16h bh = SHUF16(*(const v8h*)(WrH + kw), *(const v8h*)(WrH + kw + 8));
                v16h bl = SHUF16(*(const v8h*)(WrL + kw), *(const v8h*)(WrL + kw + 8));
                acc0 = __builtin_amdgcn_wmma_f32_16x16x32_f16(false, ah, false, bh, (short)0, acc0, false, false);
                acc0 = __builtin_amdgcn_wmma_f32_16x16x32_f16(false, ah, false, bl, (short)0, acc0, false, false);
                acc0 = __builtin_amdgcn_wmma_f32_16x16x32_f16(false, al, false, bh, (short)0, acc0, false, false);
            }
            {   // col tile 1
                const half_t* WrH = &WthL[(16 + colLane) * APADH];
                const half_t* WrL = &WtlL[(16 + colLane) * APADH];
                v16h bh = SHUF16(*(const v8h*)(WrH + kw), *(const v8h*)(WrH + kw + 8));
                v16h bl = SHUF16(*(const v8h*)(WrL + kw), *(const v8h*)(WrL + kw + 8));
                acc1 = __builtin_amdgcn_wmma_f32_16x16x32_f16(false, ah, false, bh, (short)0, acc1, false, false);
                acc1 = __builtin_amdgcn_wmma_f32_16x16x32_f16(false, ah, false, bl, (short)0, acc1, false, false);
                acc1 = __builtin_amdgcn_wmma_f32_16x16x32_f16(false, al, false, bh, (short)0, acc1, false, false);
            }
        }
        __syncthreads();
    }

    // ---- epilogue: bias + store. C VGPR r -> row (r + 8*half), col = lane&15 ----
    const float bv0 = bias[tileCol + colLane];
    const float bv1 = bias[tileCol + 16 + colLane];
    #pragma unroll
    for (int r = 0; r < 8; r++) {
        int grow = rowBase + wave * 16 + r + 8 * half;
        if (grow < M) {
            Cb[(long)grow * strideC + tileCol + colLane]      = acc0[r] + bv0;
            Cb[(long)grow * strideC + tileCol + 16 + colLane] = acc1[r] + bv1;
        }
    }
}

// One wave per (row, head). lane = head-dim channel (HD=32 == wave32).
// Each corner gather = 32 consecutive floats (128B line) -> coalesced.
__global__ __launch_bounds__(256) void msda_sample_kernel(
    const float* __restrict__ qoff,    // [NROWS, 256] (bias already added)
    const float* __restrict__ qattn,   // [NROWS, 128] (bias already added)
    const float* __restrict__ vproj,   // [B, NTOTAL, 256]
    const float* __restrict__ refpts,  // [Q, B, 2]
    const int*   __restrict__ ss,      // [4,2] (H,W)
    const int*   __restrict__ lsi,     // [4]
    float* __restrict__ agg)           // [NROWS, 256]
{
    const int wid  = blockIdx.x * WAVES + (threadIdx.x >> 5);
    const int lane = threadIdx.x & 31;
    const int h    = wid & 7;
    const int row  = wid >> 3;
    if (row >= NROWS) return;
    const int b = row & 7;              // row = qi*BATCH + b

    // --- softmax over the 16 (level,point) logits for this head ---
    const float* ap = qattn + (long)row * 128 + h * 16;
    float t[16];
    float maxv = -1e30f;
    #pragma unroll
    for (int p = 0; p < 16; p++) { t[p] = ap[p]; maxv = fmaxf(maxv, t[p]); }
    float sum = 0.f;
    #pragma unroll
    for (int p = 0; p < 16; p++) { t[p] = expf(t[p] - maxv); sum += t[p]; }
    float inv = 1.f / sum;
    #pragma unroll
    for (int p = 0; p < 16; p++) t[p] *= inv;

    const float rx = refpts[(long)row * 2 + 0];
    const float ry = refpts[(long)row * 2 + 1];
    const float* vb = vproj + (long)b * NTOTAL * D_MODEL + h * HDIM + lane;
    const float* op = qoff + (long)row * 256;

    float accum = 0.f;
    #pragma unroll
    for (int lvl = 0; lvl < NLVL; lvl++) {
        const int   Hh = ss[2 * lvl], Ww = ss[2 * lvl + 1];
        const int   start = lsi[lvl];
        const float fw = (float)Ww, fh = (float)Hh;
        #pragma unroll
        for (int p = 0; p < NPT; p++) {
            const int oidx = ((h * NLVL + lvl) * NPT + p) * 2;
            float lx = rx + op[oidx + 0] / (fw + 1e-6f);
            float ly = ry + op[oidx + 1] / (fh + 1e-6f);
            lx = fminf(fmaxf(lx, 0.f), 1.f);
            ly = fminf(fmaxf(ly, 0.f), 1.f);
            // grid_sample align_corners=False pixel coords
            const float x = lx * fw - 0.5f, y = ly * fh - 0.5f;
            const float x0 = floorf(x), y0 = floorf(y);
            const float wx1 = x - x0, wy1 = y - y0;
            const int x0i = (int)x0, y0i = (int)y0;
            const float cwx[2] = {1.f - wx1, wx1};
            const float cwy[2] = {1.f - wy1, wy1};
            float s = 0.f;
            #pragma unroll
            for (int dy = 0; dy < 2; dy++) {
                #pragma unroll
                for (int dx = 0; dx < 2; dx++) {
                    const int xi = x0i + dx, yi = y0i + dy;
                    if (xi >= 0 && xi < Ww && yi >= 0 && yi < Hh) {
                        const long vidx = (long)(start + yi * Ww + xi);
                        s += vb[vidx * D_MODEL] * (cwx[dx] * cwy[dy]);
                    }
                }
            }
            accum += s * t[lvl * NPT + p];
        }
    }
    agg[(long)row * D_MODEL + h * HDIM + lane] = accum;
}

extern "C" void kernel_launch(void* const* d_in, const int* in_sizes, int n_in,
                              void* d_out, int out_size, void* d_ws, size_t ws_size,
                              hipStream_t stream) {
    const float* query  = (const float*)d_in[0];   // [Q,B,256]
    const float* refpts = (const float*)d_in[1];   // [Q,B,2]
    const float* value  = (const float*)d_in[2];   // [N,B,256]
    const int*   ss     = (const int*)  d_in[3];   // [4,2]
    const int*   lsi    = (const int*)  d_in[4];   // [4]
    const float* W_v    = (const float*)d_in[5];   // [256,256]
    const float* b_v    = (const float*)d_in[6];
    const float* W_off  = (const float*)d_in[7];   // [256,256]
    const float* b_off  = (const float*)d_in[8];
    const float* W_attn = (const float*)d_in[9];   // [256,128]
    const float* b_attn = (const float*)d_in[10];
    const float* W_out  = (const float*)d_in[11];  // [256,256]
    const float* b_out  = (const float*)d_in[12];
    float* out = (float*)d_out;                    // [Q,B,256] -> row = qi*B+b

    // workspace carve (fp32)
    float* vproj = (float*)d_ws;                                  // [B,NTOTAL,256] 178MB
    float* qoff  = vproj + (size_t)BATCH * NTOTAL * D_MODEL;      // [NROWS,256]
    float* qattn = qoff  + (size_t)NROWS * 256;                   // [NROWS,128]
    float* agg   = qattn + (size_t)NROWS * 128;                   // [NROWS,256]

    // 1) value projection: per batch b, value[:,b,:] @ W_v + b_v -> vproj[b]
    msda_wmma_gemm_f16split<<<dim3(D_MODEL / BN, NTOTAL / BM, BATCH), 256, 0, stream>>>(
        value, (long)BATCH * D_MODEL, (long)D_MODEL,
        W_v, D_MODEL, b_v,
        vproj, (long)D_MODEL, (long)NTOTAL * D_MODEL,
        NTOTAL, D_MODEL);

    const int rowGroups = (NROWS + BM - 1) / BM;   // 63

    // 2) sampling offsets: query @ W_off + b_off
    msda_wmma_gemm_f16split<<<dim3(256 / BN, rowGroups, 1), 256, 0, stream>>>(
        query, (long)D_MODEL, 0L,
        W_off, 256, b_off,
        qoff, 256L, 0L,
        NROWS, D_MODEL);

    // 3) attention logits: query @ W_attn + b_attn
    msda_wmma_gemm_f16split<<<dim3(128 / BN, rowGroups, 1), 256, 0, stream>>>(
        query, (long)D_MODEL, 0L,
        W_attn, 128, b_attn,
        qattn, 128L, 0L,
        NROWS, D_MODEL);

    // 4) softmax + bilinear sampling + head aggregation
    msda_sample_kernel<<<dim3(NROWS * NHEAD / WAVES, 1, 1), 256, 0, stream>>>(
        qoff, qattn, vproj, refpts, ss, lsi, agg);

    // 5) output projection: agg @ W_out + b_out -> d_out
    msda_wmma_gemm_f16split<<<dim3(D_MODEL / BN, rowGroups, 1), 256, 0, stream>>>(
        agg, (long)D_MODEL, 0L,
        W_out, D_MODEL, b_out,
        out, (long)D_MODEL, 0L,
        NROWS, D_MODEL);
}
